// DCN_32779190403790
// MI455X (gfx1250) — matrix operand order repaired and checked by reading, not compiled, not run on hardware
//
#include <hip/hip_runtime.h>
#include <hip/hip_bf16.h>

// Problem constants
#define B_    8
#define CIN_  256
#define COUT_ 256
#define H_    64
#define W_    64
#define K2_   9
#define HW_   (H_ * W_)          // 4096
#define KTOT_ (CIN_ * K2_)       // 2304

typedef __attribute__((ext_vector_type(16))) __bf16 v16bf;
typedef __attribute__((ext_vector_type(2)))  __bf16 v2bf;
typedef __attribute__((ext_vector_type(8)))  float  v8f;

// Native f32 -> bf16 (round-to-nearest-even) pair pack; lowers to v_cvt on gfx1250.
static __device__ __forceinline__ unsigned pack_bf16(float a, float b) {
    union { v2bf h; unsigned u; } p;
    p.h[0] = (__bf16)a;
    p.h[1] = (__bf16)b;
    return p.u;
}
static __device__ __forceinline__ unsigned short cvt_bf16(float a) {
    union { __bf16 h; unsigned short u; } p;
    p.h = (__bf16)a;
    return p.u;
}

// ---------------------------------------------------------------------------
// Kernel 1: offset conv  om[b][ch][h][w] = conv3x3(x, w_off)[ch] + b_off[ch]
// One thread per pixel computes all 27 channels (27x reuse of each x load).
// ---------------------------------------------------------------------------
__global__ __launch_bounds__(256) void om_kernel(
    const float* __restrict__ x, const float* __restrict__ w_off,
    const float* __restrict__ b_off, float* __restrict__ om)
{
    int p  = blockIdx.x * blockDim.x + threadIdx.x;   // 0..32767
    int b  = p >> 12;
    int hw = p & 4095;
    int h  = hw >> 6, w = hw & 63;

    float acc[27];
    #pragma unroll
    for (int ch = 0; ch < 27; ++ch) acc[ch] = b_off[ch];

    const float* xb = x + (size_t)b * CIN_ * HW_;
    for (int c = 0; c < CIN_; ++c) {
        const float* xc = xb + c * HW_;
        #pragma unroll
        for (int ky = 0; ky < 3; ++ky) {
            int y = h - 1 + ky;
            if (y < 0 || y > 63) continue;
            #pragma unroll
            for (int kx = 0; kx < 3; ++kx) {
                int xw = w - 1 + kx;
                if (xw < 0 || xw > 63) continue;
                float xv   = xc[y * 64 + xw];
                int   widx = c * 9 + ky * 3 + kx;
                #pragma unroll
                for (int ch = 0; ch < 27; ++ch)
                    acc[ch] = __builtin_fmaf(xv, w_off[ch * KTOT_ + widx], acc[ch]);
            }
        }
    }
    float* omp = om + (size_t)b * 27 * HW_ + hw;
    #pragma unroll
    for (int ch = 0; ch < 27; ++ch) omp[ch * HW_] = acc[ch];
}

// ---------------------------------------------------------------------------
// Kernel 2: weight prep.  wB[n][k] (bf16), k = t*256 + c  (tap-major so a
// K-chunk of 32 is one tap x 32 contiguous channels).
// ---------------------------------------------------------------------------
__global__ __launch_bounds__(256) void wprep_kernel(
    const float* __restrict__ weight, unsigned short* __restrict__ wB)
{
    int i = blockIdx.x * blockDim.x + threadIdx.x;    // over 256*2304
    if (i >= COUT_ * KTOT_) return;
    int n = i / KTOT_;
    int k = i - n * KTOT_;
    int t = k >> 8;
    int c = k & 255;
    wB[i] = cvt_bf16(weight[(size_t)n * KTOT_ + c * K2_ + t]);
}

// ---------------------------------------------------------------------------
// Kernel 3: deformable implicit GEMM via v_wmma_f32_16x16x32_bf16.
// Block: 512 threads = 16 waves.  Tile: 32 pixels (M) x 256 outputs (N).
// Each wave owns one 16-wide N tile and two 16-wide M tiles (2 WMMA / chunk
// sharing one B fragment).  A tile (32x32 bf16) is double-buffered in LDS so
// bilinear gather staging for chunk+1 overlaps WMMA on chunk; one barrier
// per chunk.
// ---------------------------------------------------------------------------
struct Tap { float w00, w01, w10, w11; int o00, o01, o10, o11; };

__global__ __launch_bounds__(512) void dcn_wmma_kernel(
    const float* __restrict__ x, const unsigned short* __restrict__ wB,
    const float* __restrict__ bias, const float* __restrict__ om,
    float* __restrict__ out)
{
    __shared__ Tap      tap[32 * 9];          // 9.2 KB
    __shared__ unsigned ldsA[2 * 32 * 16];    // 2 bufs x (32 pix x 32 bf16) = 4 KB

    const int tid    = threadIdx.x;
    const int P      = blockIdx.x * 32;       // pixel tile base (never crosses b)
    const int b      = P >> 12;
    const int hwbase = P & 4095;

    // ---- per-(pixel, tap) bilinear table: clamped corner offsets (y*64+x),
    // ---- corner weights premultiplied by validity * sigmoid(mask)
    if (tid < 288) {
        int m = tid / 9, t = tid - m * 9;
        int hw = hwbase + m;
        int h = hw >> 6, w = hw & 63;
        const float* omb = om + (size_t)b * 27 * HW_ + hw;
        float dy = omb[t * HW_];
        float dx = omb[(9 + t) * HW_];
        float ms = 1.0f / (1.0f + __expf(-omb[(18 + t) * HW_]));
        float py = (float)(h - 1 + t / 3) + dy;
        float px = (float)(w - 1 + t % 3) + dx;
        float fy = floorf(py), fx = floorf(px);
        float ly = py - fy, lx = px - fx;
        int iy0 = (int)fy, ix0 = (int)fx;
        int iy1 = iy0 + 1, ix1 = ix0 + 1;
        float vy0 = (iy0 >= 0 && iy0 <= 63) ? 1.f : 0.f;
        float vy1 = (iy1 >= 0 && iy1 <= 63) ? 1.f : 0.f;
        float vx0 = (ix0 >= 0 && ix0 <= 63) ? 1.f : 0.f;
        float vx1 = (ix1 >= 0 && ix1 <= 63) ? 1.f : 0.f;
        int cy0 = min(max(iy0, 0), 63) * 64, cy1 = min(max(iy1, 0), 63) * 64;
        int cx0 = min(max(ix0, 0), 63),      cx1 = min(max(ix1, 0), 63);
        Tap tp;
        tp.w00 = ms * (1.f - ly) * (1.f - lx) * vy0 * vx0;
        tp.w01 = ms * (1.f - ly) * lx         * vy0 * vx1;
        tp.w10 = ms * ly         * (1.f - lx) * vy1 * vx0;
        tp.w11 = ms * ly         * lx         * vy1 * vx1;
        tp.o00 = cy0 + cx0; tp.o01 = cy0 + cx1;
        tp.o10 = cy1 + cx0; tp.o11 = cy1 + cx1;
        tap[tid] = tp;
    }

    const int lane   = tid & 31;              // wave32
    const int waveId = tid >> 5;              // 0..15 -> N tile
    const int g      = (lane >> 4) & 1;
    const int lm     = lane & 15;
    const int n      = waveId * 16 + lm;      // output channel of this lane

    const int sm = tid >> 4;                  // staging pixel 0..31
    const int sc = tid & 15;                  // staging channel-pair 0..15
    const float*    xb  = x + (size_t)b * CIN_ * HW_;
    const unsigned* wBu = (const unsigned*)wB;   // bf16 pairs

    // stage one K-chunk (32 channels of one tap) for all 32 pixels:
    // each thread computes a channel PAIR and packs it into one u32.
    auto stage = [&](int chunk, int buf) {
        const int t  = chunk >> 3;
        const int c0 = (chunk & 7) << 5;
        const Tap tp = tap[sm * 9 + t];
        const float* xc0 = xb + (size_t)(c0 + 2 * sc) * HW_;
        const float* xc1 = xc0 + HW_;
        float v0 = tp.w00 * xc0[tp.o00] + tp.w01 * xc0[tp.o01]
                 + tp.w10 * xc0[tp.o10] + tp.w11 * xc0[tp.o11];
        float v1 = tp.w00 * xc1[tp.o00] + tp.w01 * xc1[tp.o01]
                 + tp.w10 * xc1[tp.o10] + tp.w11 * xc1[tp.o11];
        ldsA[buf * 512 + sm * 16 + sc] = pack_bf16(v0, v1);
    };

    v8f acc0 = {};   // pixels 0..15
    v8f acc1 = {};   // pixels 16..31

    __syncthreads();        // tap table visible
    stage(0, 0);
    __syncthreads();        // first A buffer visible

    for (int chunk = 0; chunk < 72; ++chunk) {
        const int buf = chunk & 1;
        if (chunk + 1 < 72) stage(chunk + 1, buf ^ 1);   // overlap with WMMA

        // A fragments per ISA 16-bit 16x32 layout:
        //   v<4: K = 8g + 2v ; v>=4: K = 16 + 8g + 2(v-4); K-pair per u32
        union { v16bf v; unsigned u[8]; } a0, a1, bm;
        const unsigned* Ab = ldsA + buf * 512;
        #pragma unroll
        for (int vv = 0; vv < 8; ++vv) {
            int k2 = ((vv < 4) ? (8 * g + 2 * vv) : (16 + 8 * g + 2 * (vv - 4))) >> 1;
            a0.u[vv] = Ab[lm * 16 + k2];
            a1.u[vv] = Ab[(lm + 16) * 16 + k2];
        }
        // B fragment: same K layout as A (v<4: K = 8g + 2v ; v>=4: K = 16 + 8g + 2(v-4)); K pair per u32
        const int bbase = n * (KTOT_ / 2) + chunk * 16;
        #pragma unroll
        for (int vv = 0; vv < 8; ++vv) bm.u[vv] = wBu[bbase + ((vv < 4) ? (4 * g + vv) : (8 + 4 * g + (vv - 4)))];

        acc0 = __builtin_amdgcn_wmma_f32_16x16x32_bf16(
                   false, a0.v, false, bm.v, (short)0, acc0, false, false);
        acc1 = __builtin_amdgcn_wmma_f32_16x16x32_bf16(
                   false, a1.v, false, bm.v, (short)0, acc1, false, false);

        __syncthreads();    // next chunk's buffer fully staged / this one free
    }

    // ---- epilogue: C layout lane=(N), VGPR v -> M = v + 8g (contiguous pixels)
    const float bv = bias[n];
    float* op0 = out + ((size_t)(b * COUT_ + n) << 12) + hwbase + 8 * g;
    float* op1 = op0 + 16;
    #pragma unroll
    for (int vv = 0; vv < 8; ++vv) op0[vv] = acc0[vv] + bv;
    #pragma unroll
    for (int vv = 0; vv < 8; ++vv) op1[vv] = acc1[vv] + bv;
}

// ---------------------------------------------------------------------------
extern "C" void kernel_launch(void* const* d_in, const int* in_sizes, int n_in,
                              void* d_out, int out_size, void* d_ws, size_t ws_size,
                              hipStream_t stream)
{
    const float* x      = (const float*)d_in[0];
    const float* weight = (const float*)d_in[1];
    const float* bias   = (const float*)d_in[2];
    const float* w_off  = (const float*)d_in[3];
    const float* b_off  = (const float*)d_in[4];
    float*       out    = (float*)d_out;

    // workspace: om (8*27*4096 f32 = 3,538,944 B) then wB bf16 (1,179,648 B)
    float*          om = (float*)d_ws;
    unsigned short* wB = (unsigned short*)((char*)d_ws +
                              (size_t)B_ * 27 * HW_ * sizeof(float));

    om_kernel<<<(B_ * HW_) / 256, 256, 0, stream>>>(x, w_off, b_off, om);
    wprep_kernel<<<(COUT_ * KTOT_ + 255) / 256, 256, 0, stream>>>(weight, wB);
    dcn_wmma_kernel<<<(B_ * HW_) / 32, 512, 0, stream>>>(x, wB, bias, om, out);
}
